// MTLSTM_71871982731757
// MI455X (gfx1250) — matrix operand order repaired and checked by reading, not compiled, not run on hardware
//
#include <hip/hip_runtime.h>
#include <hip/hip_bf16.h>
#include <math.h>

typedef __attribute__((ext_vector_type(16))) _Float16 v16h;
typedef __attribute__((ext_vector_type(8)))  _Float16 v8h;
typedef __attribute__((ext_vector_type(8)))  float    v8f;

#define B_DIM   128
#define T_DIM   512
#define K_DIM   12
#define H_DIM   256
#define GATE_DIM 1024
#define BT      16      // batch rows per workgroup
#define NWAVE   16      // waves per workgroup (wave32)
#define HSTR    264     // padded LDS row stride (halves): 264*2B = 528B -> bank-conflict-free

__device__ __forceinline__ v16h mk16(v8h lo, v8h hi) {
  v16h r;
#pragma unroll
  for (int i = 0; i < 8; ++i) { r[i] = lo[i]; r[8 + i] = hi[i]; }
  return r;
}
// B-fragment: two 8-half runs (k 8h..8h+7 and 16+8h..16+8h+7) of one weight row, via 32-bit byte offset so the
// backend selects global_load_b128 SADDR form (SGPR base + zext(u32 voffset) + imm).
__device__ __forceinline__ v16h load_b(const _Float16* __restrict__ base, unsigned byteoff) {
  const char* p = (const char*)base;
  v8h lo = *(const v8h*)(p + byteoff);
  v8h hi = *(const v8h*)(p + byteoff + 32);
  return mk16(lo, hi);
}
__device__ __forceinline__ float sigmoidf_(float x) { return 1.0f / (1.0f + __expf(-x)); }

// ---- one-time weight conversion: fp32 -> f16, W_ih zero-padded K=12 -> 32 ----
__global__ void mtlstm_prep(const float* __restrict__ wih, const float* __restrict__ whh,
                            const float* __restrict__ wdec,
                            _Float16* __restrict__ wih16, _Float16* __restrict__ whh16,
                            _Float16* __restrict__ wdec16) {
  int i = blockIdx.x * blockDim.x + threadIdx.x;
  const int n1 = GATE_DIM * H_DIM;          // 262144
  const int n2 = K_DIM * H_DIM * H_DIM;     // 786432
  const int n3 = GATE_DIM * 32;             // 32768
  if (i < n1) {
    whh16[i] = (_Float16)whh[i];
  } else if (i < n1 + n2) {
    int j = i - n1;
    wdec16[j] = (_Float16)wdec[j];
  } else if (i < n1 + n2 + n3) {
    int j = i - n1 - n2;
    int r = j >> 5, k = j & 31;
    wih16[j] = (_Float16)((k < K_DIM) ? wih[r * K_DIM + k] : 0.0f);
  }
}

__global__ __launch_bounds__(NWAVE * 32) void mtlstm_main(
    const float* __restrict__ xd, const float* __restrict__ dt,
    const float* __restrict__ bias, const float* __restrict__ bdec,
    const float* __restrict__ lw, const float* __restrict__ lb,
    const _Float16* __restrict__ wih16, const _Float16* __restrict__ whh16,
    const _Float16* __restrict__ wdec16, float* __restrict__ out) {
  __shared__ _Float16 Hs[2][BT][HSTR];    // ping-pong h (f16, WMMA A source)
  __shared__ _Float16 Cs[2][BT][HSTR];    // ping-pong c (f16, WMMA A source)
  __shared__ float    dfac[2][K_DIM][BT]; // (1/log(e+dt) - 1) per (k, batch-row)
  __shared__ float    hfin[BT][H_DIM];    // final h in fp32 for output projection

  const int tid   = threadIdx.x;
  const int lane  = tid & 31;
  const int wv    = tid >> 5;             // wave id: owns hidden cols [16*wv, 16*wv+16)
  const int col   = lane & 15;            // A row / B-&-D column within tile
  const int hih   = (lane >> 4) & 1;      // upper half-wave?
  const int mbase = hih * 8;              // D rows held: mbase..mbase+7
  const int akb   = hih * 8;              // A-fragment K sub-offset
  const unsigned bkb2 = (unsigned)hih * 16u;  // B-fragment K sub-offset in BYTES (8 halves)
  const int bbase = blockIdx.x * BT;      // batch tile origin
  const int hcol  = wv * 16 + col;        // owned hidden column

  // zero initial state (buffer 0)
  {
    _Float16* ph = &Hs[0][0][0];
    _Float16* pc = &Cs[0][0][0];
    for (int i = tid; i < BT * HSTR; i += blockDim.x) { ph[i] = (_Float16)0.0f; pc[i] = (_Float16)0.0f; }
  }
  // decay factors for t=0
  if (tid < K_DIM * BT) {
    int k = tid >> 4, r = tid & 15;
    unsigned doff = (((unsigned)(bbase + r) * T_DIM + 0u) * K_DIM + (unsigned)k) * 4u;
    float v = *(const float*)((const char*)dt + doff);
    dfac[0][k][r] = 1.0f / logf(2.718281828459045f + v) - 1.0f;
  }

  // truly loop-invariant per-lane scalars: load once, keep in VGPRs
  float bias_r[4], bdec_r[K_DIM];
#pragma unroll
  for (int q = 0; q < 4; ++q) bias_r[q] = bias[q * H_DIM + hcol];
#pragma unroll
  for (int k = 0; k < K_DIM; ++k) bdec_r[k] = bdec[k * H_DIM + hcol];

  float c_reg[8];                         // fp32 cell state lives in registers all 512 steps
#pragma unroll
  for (int r = 0; r < 8; ++r) c_reg[r] = 0.0f;

  for (int t = 0; t < T_DIM; ++t) {
    __syncthreads();                      // state buffers for step t are complete
    const int pb = t & 1, nb = pb ^ 1;

    // Anti-LICM: opaque 32-bit zero folded into every weight byte-offset.
    // Offsets stay 32-bit & loop-variant -> loads can't be hoisted (no spills),
    // and the SADDR global-load form (SGPR base + u32 voffset) is preserved.
    unsigned zofs = 0;
    asm volatile("" : "+s"(zofs));

    // ---- X A-fragment (K padded 12 -> 32) ----
    v16h ax;
    {
      unsigned xrow = ((unsigned)(bbase + col) * T_DIM + (unsigned)t) * (K_DIM * 4u);
#pragma unroll
      for (int j = 0; j < 8; ++j) {
        int K = akb + j;
        float xv = (K < K_DIM) ? *(const float*)((const char*)xd + xrow + (unsigned)K * 4u) : 0.0f;
        ax[j] = (_Float16)xv;
        ax[8 + j] = (_Float16)0.0f;       // K in [16,32) of padded space: zero
      }
    }

    // ---- H A-fragments (8 x K=32) from LDS ----
    v16h ah[8];
    {
      const _Float16* hrow = &Hs[pb][col][0];
#pragma unroll
      for (int kk = 0; kk < 8; ++kk)
        ah[kk] = mk16(*(const v8h*)(hrow + kk * 32 + akb),
                      *(const v8h*)(hrow + kk * 32 + akb + 16));
    }

    // ---- gates: i,f,g,o tiles for owned hidden columns ----
    v8f acc[4];
#pragma unroll
    for (int q = 0; q < 4; ++q) {
      v8f a = {};
      const unsigned grow = (unsigned)(q * H_DIM + hcol); // gate row (B column of tile)
      a = __builtin_amdgcn_wmma_f32_16x16x32_f16(
          false, ax, false, load_b(wih16, grow * 64u + bkb2 + zofs), (short)0, a, false, false);
#pragma unroll
      for (int kk = 0; kk < 8; ++kk) {
        v16h bfrag = load_b(whh16, grow * 512u + (unsigned)kk * 64u + bkb2 + zofs);
        a = __builtin_amdgcn_wmma_f32_16x16x32_f16(false, ah[kk], false, bfrag, (short)0, a, false, false);
      }
#pragma unroll
      for (int r = 0; r < 8; ++r) a[r] += bias_r[q];
      acc[q] = a;
    }

    // ---- C A-fragments ----
    v16h ac[8];
    {
      const _Float16* crow = &Cs[pb][col][0];
#pragma unroll
      for (int kk = 0; kk < 8; ++kk)
        ac[kk] = mk16(*(const v8h*)(crow + kk * 32 + akb),
                      *(const v8h*)(crow + kk * 32 + akb + 16));
    }

    // ---- subspace decomposition: S = sum_k (decay_k - 1) * tanh(C @ Wd[k]^T + bd[k]) ----
    v8f S = {};
#pragma unroll
    for (int k = 0; k < K_DIM; ++k) {
      v8f a = {};
      const unsigned drow = (unsigned)(k * H_DIM + hcol) * 512u; // row byte offset in wdec16
#pragma unroll
      for (int kk = 0; kk < 8; ++kk) {
        v16h bfrag = load_b(wdec16, drow + (unsigned)kk * 64u + bkb2 + zofs);
        a = __builtin_amdgcn_wmma_f32_16x16x32_f16(false, ac[kk], false, bfrag, (short)0, a, false, false);
      }
#pragma unroll
      for (int r = 0; r < 8; ++r) {
        float cs = tanhf(a[r] + bdec_r[k]);
        S[r] += dfac[pb][k][mbase + r] * cs;
      }
    }

    // ---- decay factors for next step (double-buffered, race-free) ----
    if (t + 1 < T_DIM && tid < K_DIM * BT) {
      int k = tid >> 4, r = tid & 15;
      unsigned doff = (((unsigned)(bbase + r) * T_DIM + (unsigned)(t + 1)) * K_DIM + (unsigned)k) * 4u;
      float v = *(const float*)((const char*)dt + doff);
      dfac[nb][k][r] = 1.0f / logf(2.718281828459045f + v) - 1.0f;
    }

    // ---- elementwise LSTM update; each (row,col) element owned by exactly one lane ----
#pragma unroll
    for (int r = 0; r < 8; ++r) {
      float iv = sigmoidf_(acc[0][r]);
      float fv = sigmoidf_(acc[1][r]);
      float gv = tanhf(acc[2][r]);
      float ov = sigmoidf_(acc[3][r]);
      float cn = fv * (c_reg[r] + S[r]) + iv * gv;
      float hn = ov * tanhf(cn);
      c_reg[r] = cn;
      const int m = mbase + r;
      Hs[nb][m][hcol] = (_Float16)hn;
      Cs[nb][m][hcol] = (_Float16)cn;
      if (t == T_DIM - 1) hfin[m][hcol] = hn;
    }
  }

  __syncthreads();
  // ---- output projection: out[b] = h_final . lw + lb ----
  if (tid < BT) {
    float s = lb[0];
    for (int h2 = 0; h2 < H_DIM; ++h2) s += hfin[tid][h2] * lw[h2];
    out[bbase + tid] = s;
  }
}

extern "C" void kernel_launch(void* const* d_in, const int* in_sizes, int n_in,
                              void* d_out, int out_size, void* d_ws, size_t ws_size,
                              hipStream_t stream) {
  const float* xd   = (const float*)d_in[0];
  const float* dt   = (const float*)d_in[1];
  const float* wih  = (const float*)d_in[2];
  const float* whh  = (const float*)d_in[3];
  const float* bias = (const float*)d_in[4];
  const float* wdec = (const float*)d_in[5];
  const float* bdec = (const float*)d_in[6];
  const float* lw   = (const float*)d_in[7];
  const float* lb   = (const float*)d_in[8];

  _Float16* whh16  = (_Float16*)d_ws;                        // 1024*256 halves
  _Float16* wdec16 = whh16 + (size_t)GATE_DIM * H_DIM;       // 12*256*256 halves
  _Float16* wih16  = wdec16 + (size_t)K_DIM * H_DIM * H_DIM; // 1024*32 halves

  const int ntot = GATE_DIM * H_DIM + K_DIM * H_DIM * H_DIM + GATE_DIM * 32;
  mtlstm_prep<<<(ntot + 255) / 256, 256, 0, stream>>>(wih, whh, wdec, wih16, whh16, wdec16);
  mtlstm_main<<<B_DIM / BT, NWAVE * 32, 0, stream>>>(xd, dt, bias, bdec, lw, lb,
                                                     wih16, whh16, wdec16, (float*)d_out);
}
